// LocalAttention_10299331575842
// MI455X (gfx1250) — matrix operand order repaired and checked by reading, not compiled, not run on hardware
//
#include <hip/hip_runtime.h>
#include <hip/hip_bf16.h>

typedef __attribute__((ext_vector_type(16))) _Float16 v16h;
typedef __attribute__((ext_vector_type(8)))  float    v8f;
typedef __attribute__((ext_vector_type(4)))  int      i32x4;

#define S_LEN 4096
#define EMB   256
#define NH    8
#define HD    32
#define WIN   64

// ---------------- CDNA5 async global->LDS copy (guarded) ----------------

#if __has_builtin(__builtin_amdgcn_global_load_async_to_lds_b128) && \
    __has_builtin(__builtin_amdgcn_s_wait_asynccnt)
#define USE_ASYNC_LDS 1
#else
#define USE_ASYNC_LDS 0
#endif

__device__ inline void cp_b128(void* lds_dst, const void* gsrc) {
#if USE_ASYNC_LDS
  __builtin_amdgcn_global_load_async_to_lds_b128(
      (__attribute__((address_space(1))) i32x4*)gsrc,
      (__attribute__((address_space(3))) i32x4*)lds_dst, 0, 0);
#else
  *(uint4*)lds_dst = *(const uint4*)gsrc;
#endif
}

__device__ inline void wait_async() {
#if USE_ASYNC_LDS
  __builtin_amdgcn_s_wait_asynccnt(0);
#endif
}

// ---------------- WMMA helpers ----------------

__device__ inline v8f wmma_f16(v16h a, v16h b, v8f c) {
  return __builtin_amdgcn_wmma_f32_16x16x32_f16(false, a, false, b, (short)0, c,
                                                false, false);
}

// A-matrix fragment (16xK tile, 16-bit). Lane holds row m = lane&15.
// halfs 0..15 = src[16*g .. +15]
__device__ inline v16h ldA(const _Float16* p, int g) {
  union { uint4 u[2]; v16h h; } u;
  const uint4* q = (const uint4*)(p + 16 * g);
  u.u[0] = q[0];
  u.u[1] = q[1];
  return u.h;
}

// B-matrix fragment (Kx16) when the SOURCE is row-major [N x K] (we multiply
// by its transpose). Lane holds column n = lane&15 (one source row).
// halfs 0..15 = srcrow[16*g .. +15]  (contiguous 32 bytes)
__device__ inline v16h ldB_rows(const _Float16* p, int g) {
  union { uint4 u[2]; v16h h; } u;
  const uint4* q = (const uint4*)(p + 16 * g);
  u.u[0] = q[0];
  u.u[1] = q[1];
  return u.h;
}

// B-matrix fragment when the SOURCE is row-major [K x N] (true B orientation).
// p points at element (k=0, n=lane&15); stride = row stride in elements.
__device__ inline v16h ldB_cols(const _Float16* p, int g, int stride) {
  v16h r;
  const _Float16* q = p + 16 * g * stride;
#pragma unroll
  for (int j = 0; j < 16; ++j) r[j] = q[j * stride];
  return r;
}

// ---------------- kernels ----------------

__global__ void cvt_f32_f16(const float* __restrict__ in,
                            _Float16* __restrict__ out, int n) {
  int i = blockIdx.x * blockDim.x + threadIdx.x;
  if (i < n) out[i] = (_Float16)in[i];
}

// C[M,N] = A[M,K] * B[N,K]^T + bias.  Block = 8 waves = 128x64 tile; the
// 64x256 B-tile is staged once in LDS (async) and shared by all waves.
__global__ __launch_bounds__(256) void gemm_wmma(
    const _Float16* __restrict__ A, const _Float16* __restrict__ Bw,
    const float* __restrict__ bias, _Float16* __restrict__ out_h,
    float* __restrict__ out_f, int N) {
  constexpr int K = 256;
  __shared__ __align__(16) _Float16 Bl[64 * K];   // 32 KB

  int tid = threadIdx.x;
  int lane = tid & 31, wave = tid >> 5;
  int r = lane & 15, g = lane >> 4;
  int m0 = blockIdx.x * 128 + wave * 16;
  int n0 = blockIdx.y * 64;

  // stage B tile: 64 rows x 512 bytes = 2048 16B chunks, 8 per thread
#pragma unroll
  for (int i = 0; i < 8; ++i) {
    int idx = tid + i * 256;
    int row = idx >> 5, c = idx & 31;
    cp_b128(&Bl[row * K + c * 8], Bw + (size_t)(n0 + row) * K + c * 8);
  }
  wait_async();
  __syncthreads();

  v8f acc[4] = {v8f{}, v8f{}, v8f{}, v8f{}};
  const _Float16* arow = A + (size_t)(m0 + r) * K;
#pragma unroll
  for (int kc = 0; kc < 8; ++kc) {                // K = 256 = 8 * 32
    v16h a = ldA(arow + kc * 32, g);
#pragma unroll
    for (int t = 0; t < 4; ++t) {
      v16h b = ldB_rows(&Bl[(t * 16 + r) * K + kc * 32], g);
      acc[t] = wmma_f16(a, b, acc[t]);
    }
  }
#pragma unroll
  for (int t = 0; t < 4; ++t) {
    int f = n0 + t * 16 + r;
    float bb = bias[f];
#pragma unroll
    for (int e = 0; e < 8; ++e) {
      int m = m0 + e + 8 * g;
      float v = acc[t][e] + bb;
      if (out_h) out_h[(size_t)m * N + f] = (_Float16)v;
      else       out_f[(size_t)m * N + f] = v;
    }
  }
}

// Banded attention: one wave per (b, h, 16-query tile).
__global__ __launch_bounds__(32) void attn_band(
    const _Float16* __restrict__ qkv, _Float16* __restrict__ attn_out) {
  __shared__ __align__(16) _Float16 Pl[16 * 160];   // probs, row-major
  __shared__ __align__(16) _Float16 Vl[160 * 32];   // V rows (key x d)

  int lane = threadIdx.x & 31;
  int r = lane & 15, g = lane >> 4;
  int tile = blockIdx.x & 255;            // S/16 = 256
  int bh = blockIdx.x >> 8;
  int b = bh >> 3, h = bh & 7;
  int q0 = tile * 16;
  int klo = q0 - WIN;  if (klo < 0) klo = 0;
  int khi = q0 + 16 + WIN; if (khi > S_LEN) khi = S_LEN;
  int nt = (khi - klo) >> 4;              // 5..9 whole key tiles

  const size_t LDR = 3 * EMB;             // qkv row stride (768)
  const _Float16* base = qkv + (size_t)b * S_LEN * LDR;

  // ---- stage V rows to LDS early (async, overlapped with scores) ----
  int nrows = nt * 16;
  for (int i = lane; i < nrows * 4; i += 32) {
    int row = i >> 2, c4 = i & 3;
    cp_b128(&Vl[row * 32 + c4 * 8],
            base + (size_t)(klo + row) * LDR + 2 * EMB + h * HD + c4 * 8);
  }
  int nc = (nt + 1) >> 1;                 // K=32 chunks for P*V
  for (int i = lane; i < (nc * 32 - nrows) * 4; i += 32) {  // zero pad rows
    int row = nrows + (i >> 2), c4 = i & 3;
    uint4 z4 = {0u, 0u, 0u, 0u};
    *(uint4*)&Vl[row * 32 + c4 * 8] = z4;
  }

  // Q fragment (A layout), head slice is 32 contiguous halfs
  v16h aQ = ldA(base + (size_t)(q0 + r) * LDR + h * HD, g);

  // ---- scores: up to 9 WMMAs ----
  v8f sc[9];
#pragma unroll
  for (int t = 0; t < 9; ++t) {
    if (t < nt) {
      int kb = klo + t * 16;
      v16h bK = ldB_rows(base + (size_t)(kb + r) * LDR + EMB + h * HD, g);
      v8f z = {};
      sc[t] = wmma_f16(aQ, bK, z);
    } else {
      sc[t] = v8f{};
    }
  }

  // ---- scale + band mask + row max ----
  const float scale = 0.17677669529663687f;   // 1/sqrt(32)
  float lm[8], ls[8];
#pragma unroll
  for (int e = 0; e < 8; ++e) lm[e] = -3.0e38f;
#pragma unroll
  for (int t = 0; t < 9; ++t) {
    int kb = klo + t * 16;
#pragma unroll
    for (int e = 0; e < 8; ++e) {
      int qi = q0 + e + 8 * g;
      int ki = kb + r;
      int d = qi - ki;
      float v = (t < nt && d <= WIN && d >= -WIN) ? sc[t][e] * scale : -1.0e30f;
      sc[t][e] = v;
      lm[e] = fmaxf(lm[e], v);
    }
  }
#pragma unroll
  for (int e = 0; e < 8; ++e) {
    float m = lm[e];
    m = fmaxf(m, __shfl_xor(m, 1, 32));
    m = fmaxf(m, __shfl_xor(m, 2, 32));
    m = fmaxf(m, __shfl_xor(m, 4, 32));
    m = fmaxf(m, __shfl_xor(m, 8, 32));
    lm[e] = m;
    ls[e] = 0.0f;
  }
#pragma unroll
  for (int t = 0; t < 9; ++t)
#pragma unroll
    for (int e = 0; e < 8; ++e) {
      float p = __expf(sc[t][e] - lm[e]);
      sc[t][e] = p;
      ls[e] += p;
    }
#pragma unroll
  for (int e = 0; e < 8; ++e) {
    float s = ls[e];
    s += __shfl_xor(s, 1, 32);
    s += __shfl_xor(s, 2, 32);
    s += __shfl_xor(s, 4, 32);
    s += __shfl_xor(s, 8, 32);
    ls[e] = 1.0f / s;
  }

  // ---- P to LDS (f16, rows 16 x 160) ----
#pragma unroll
  for (int t = 0; t < 9; ++t)
#pragma unroll
    for (int e = 0; e < 8; ++e)
      Pl[(e + 8 * g) * 160 + t * 16 + r] = (_Float16)sc[t][e];
  {  // zero columns 144..159 (beyond the 9 tiles)
    uint4 z4 = {0u, 0u, 0u, 0u};
    *(uint4*)&Pl[r * 160 + 144 + 8 * g] = z4;
  }

  wait_async();
  __syncthreads();

  // ---- out = P * V : up to 10 WMMAs ----
  v8f acc0 = {}, acc1 = {};
#pragma unroll
  for (int c = 0; c < 5; ++c) {
    if (c < nc) {
      v16h aP = ldA(Pl + r * 160 + c * 32, g);
      v16h b0 = ldB_cols(Vl + c * 32 * 32 + r, g, 32);
      v16h b1 = ldB_cols(Vl + c * 32 * 32 + 16 + r, g, 32);
      acc0 = wmma_f16(aP, b0, acc0);
      acc1 = wmma_f16(aP, b1, acc1);
    }
  }

  // ---- normalize by row sum and store f16 [B,S,E] ----
  _Float16* orow = attn_out + ((size_t)b * S_LEN + q0) * EMB + h * HD;
#pragma unroll
  for (int e = 0; e < 8; ++e) {
    orow[(size_t)(e + 8 * g) * EMB + r]      = (_Float16)(acc0[e] * ls[e]);
    orow[(size_t)(e + 8 * g) * EMB + 16 + r] = (_Float16)(acc1[e] * ls[e]);
  }
}

// ---------------- launcher ----------------

extern "C" void kernel_launch(void* const* d_in, const int* in_sizes, int n_in,
                              void* d_out, int out_size, void* d_ws,
                              size_t ws_size, hipStream_t stream) {
  const float* x      = (const float*)d_in[0];
  const float* w_in   = (const float*)d_in[1];
  const float* b_in   = (const float*)d_in[2];
  const float* w_out  = (const float*)d_in[3];
  const float* b_out  = (const float*)d_in[4];
  float* out = (float*)d_out;

  const int B = 2;
  const int M = B * S_LEN;                 // 8192 rows
  const int nx  = M * EMB;                 // x elements
  const int nw1 = 3 * EMB * EMB;           // in_proj_w
  const int nw2 = EMB * EMB;               // out_w

  _Float16* ws     = (_Float16*)d_ws;
  _Float16* xh     = ws;
  _Float16* wqkvh  = xh + nx;
  _Float16* wouth  = wqkvh + nw1;
  _Float16* qkvh   = wouth + nw2;
  _Float16* attnh  = qkvh + (size_t)M * 3 * EMB;

  cvt_f32_f16<<<(nx  + 255) / 256, 256, 0, stream>>>(x,     xh,    nx);
  cvt_f32_f16<<<(nw1 + 255) / 256, 256, 0, stream>>>(w_in,  wqkvh, nw1);
  cvt_f32_f16<<<(nw2 + 255) / 256, 256, 0, stream>>>(w_out, wouth, nw2);

  // qkv = x @ Wqkv^T + b  (M x 768)
  dim3 g1(M / 128, (3 * EMB) / 64);
  gemm_wmma<<<g1, 256, 0, stream>>>(xh, wqkvh, b_in, qkvh, nullptr, 3 * EMB);

  // banded attention
  attn_band<<<B * NH * (S_LEN / 16), 32, 0, stream>>>(qkvh, attnh);

  // y = attn @ Wout^T + b  (M x 256, f32 out)
  dim3 g2(M / 128, EMB / 64);
  gemm_wmma<<<g2, 256, 0, stream>>>(attnh, wouth, b_out, nullptr, out, EMB);
}
